// MultiHead_32856499814545
// MI455X (gfx1250) — matrix operand order repaired and checked by reading, not compiled, not run on hardware
//
#include <hip/hip_runtime.h>
#include <stdint.h>

// Problem constants (match reference)
#define B_ 4
#define T_ 2048
#define E_ 1024
#define H_ 16
#define D_ 64

typedef __attribute__((ext_vector_type(16))) __bf16   v16bf;
typedef __attribute__((ext_vector_type(8)))  float    v8f;
typedef __attribute__((ext_vector_type(4)))  unsigned int u32x4;
typedef __attribute__((ext_vector_type(4)))  float    f32x4;

#if defined(__HIP_DEVICE_COMPILE__) && __has_builtin(__builtin_amdgcn_global_load_async_to_lds_b128)
#define HAVE_ASYNC_LDS 1
// Prototype (from hipcc diagnostic): param0 = 'vector_size(16) int __device__ *'
typedef int v4i_vs __attribute__((vector_size(16)));
typedef __attribute__((address_space(1))) v4i_vs* as1_v4i_ptr;   // __device__
typedef __attribute__((address_space(3))) v4i_vs* as3_v4i_ptr;   // __shared__
#else
#define HAVE_ASYNC_LDS 0
#endif

union FragBf {
    v16bf v;
    u32x4 q[2];
    unsigned int u[8];
    unsigned short h[16];
};

// f32 pair -> packed bf16 dword.  CDNA5 keeps the CDNA4 v_cvt_pk_bf16_f32
// (bf16 VALU datapath is present: V_PK_*_BF16, V_FMA_MIX*_BF16) — emit it
// directly; the clang builtin is not exposed for gfx1250.
__device__ __forceinline__ unsigned int pack_bf2(float a, float b) {
#if defined(__HIP_DEVICE_COMPILE__)
    unsigned int r;
    asm("v_cvt_pk_bf16_f32 %0, %1, %2" : "=v"(r) : "v"(a), "v"(b));
    return r;
#else
    unsigned int ua = __float_as_uint(a);
    ua += 0x7FFFu + ((ua >> 16) & 1u);
    unsigned int ub = __float_as_uint(b);
    ub += 0x7FFFu + ((ub >> 16) & 1u);
    return (ua >> 16) | (ub & 0xFFFF0000u);
#endif
}
__device__ __forceinline__ unsigned short f2bf(float f) {
    return (unsigned short)(pack_bf2(f, f) & 0xFFFFu);
}
__device__ __forceinline__ v8f vzero8() {
    v8f z; for (int i = 0; i < 8; ++i) z[i] = 0.0f; return z;
}

// ---------------------------------------------------------------------------
// Kernel 1: per-head QKV projection.  C[16t x 64d] = data[16t x 1024e] * W[e,d]
// Block = 128 threads (4 waves); wave w handles rows t0..t0+15.
// 64-deep K chunks per barrier period (8 WMMAs between syncs).
// grid.x = 3 * B * H * (T/64)
// ---------------------------------------------------------------------------
__global__ __launch_bounds__(128) void qkv_proj_kernel(
    const float* __restrict__ data,
    const float* __restrict__ Wq, const float* __restrict__ Wk,
    const float* __restrict__ Wv,
    unsigned short* __restrict__ Qb, unsigned short* __restrict__ Kb,
    unsigned short* __restrict__ Vb)
{
    const int tid  = threadIdx.x;
    const int lane = tid & 31;
    const int wave = tid >> 5;
    const int hi   = lane >> 4;
    const int lx   = lane & 15;

    int x = blockIdx.x;
    const int tchunk = x & 31;  x >>= 5;          // T/64 = 32
    const int h      = x & 15;  x >>= 4;          // H = 16
    const int b      = x & 3;   x >>= 2;          // B = 4
    const int qkv    = x;                          // 0,1,2

    const float* W = (qkv == 0) ? Wq : (qkv == 1) ? Wk : Wv;
    W += (size_t)h * E_ * D_;
    unsigned short* dst = (qkv == 0) ? Qb : (qkv == 1) ? Kb : Vb;

    const int t0 = tchunk * 64 + wave * 16;
    const float* Arow = data + ((size_t)b * T_ + t0) * E_;

    __shared__ alignas(16) unsigned short Wt[64 * 64];   // [d][e_local] bf16, 8KB

    v8f acc[4];
    for (int nb = 0; nb < 4; ++nb) acc[nb] = vzero8();

    for (int e0 = 0; e0 < E_; e0 += 64) {
        if (e0 + 64 < E_) __builtin_prefetch(W + (size_t)(e0 + 64) * D_, 0, 0);

        // Cooperative load of W[e0..e0+63][0..63] -> Wt[d][e] (bf16, packed)
        #pragma unroll
        for (int r = 0; r < 16; ++r) {
            int idx  = tid + 128 * r;     // 0..2047 = d*32 + epair
            int dcol = idx >> 5;
            int ep   = idx & 31;
            float w0 = W[(size_t)(e0 + 2 * ep)     * D_ + dcol];
            float w1 = W[(size_t)(e0 + 2 * ep + 1) * D_ + dcol];
            *(unsigned int*)&Wt[dcol * 64 + 2 * ep] = pack_bf2(w0, w1);
        }
        __syncthreads();

        // Two A fragments (16x32 bf16 each) from f32 data rows
        FragBf a[2];
        #pragma unroll
        for (int half = 0; half < 2; ++half) {
            const int be = e0 + half * 32 + (hi ? 8 : 0);
            const float* p = Arow + (size_t)lx * E_ + be;
            f32x4 f0 = *(const f32x4*)(p);
            f32x4 f1 = *(const f32x4*)(p + 4);
            f32x4 f2 = *(const f32x4*)(p + 16);
            f32x4 f3 = *(const f32x4*)(p + 20);
            a[half].u[0] = pack_bf2(f0.x, f0.y); a[half].u[1] = pack_bf2(f0.z, f0.w);
            a[half].u[2] = pack_bf2(f1.x, f1.y); a[half].u[3] = pack_bf2(f1.z, f1.w);
            a[half].u[4] = pack_bf2(f2.x, f2.y); a[half].u[5] = pack_bf2(f2.z, f2.w);
            a[half].u[6] = pack_bf2(f3.x, f3.y); a[half].u[7] = pack_bf2(f3.z, f3.w);
        }

        // Load ALL B fragments first (distinct registers -> DS loads pipeline),
        // then issue the 8 WMMAs back-to-back.
        FragBf bf[8];
        #pragma unroll
        for (int half = 0; half < 2; ++half)
            #pragma unroll
            for (int nb = 0; nb < 4; ++nb) {
                const unsigned short* p =
                    &Wt[(nb * 16 + lx) * 64 + half * 32 + (hi ? 8 : 0)];
                bf[half * 4 + nb].q[0] = *(const u32x4*)(p);
                bf[half * 4 + nb].q[1] = *(const u32x4*)(p + 16);
            }
        #pragma unroll
        for (int half = 0; half < 2; ++half)
            #pragma unroll
            for (int nb = 0; nb < 4; ++nb)
                acc[nb] = __builtin_amdgcn_wmma_f32_16x16x32_bf16(
                    false, a[half].v, false, bf[half * 4 + nb].v,
                    (short)0, acc[nb], false, false);
        __syncthreads();
    }

    // Store bf16 [B,H,T,D]; Q pre-scaled by 1/sqrt(D)
    const float scale = (qkv == 0) ? 0.125f : 1.0f;
    unsigned short* dbase = dst + (((size_t)b * H_ + h) * T_) * D_;
    #pragma unroll
    for (int nb = 0; nb < 4; ++nb)
        #pragma unroll
        for (int v = 0; v < 8; ++v) {
            int t = t0 + v + 8 * hi;
            int d = nb * 16 + lx;
            dbase[(size_t)t * D_ + d] = f2bf(acc[nb][v] * scale);
        }
}

// ---------------------------------------------------------------------------
// Kernel 2: causal flash attention, one wave per 16-query block.
// K tiles staged through LDS with GLOBAL_LOAD_ASYNC_TO_LDS_B128, overlapped
// with the V-transpose work; V transposed in LDS.
// grid.x = B * H * (T/16);  block = 32 threads (one wave)
// ---------------------------------------------------------------------------
__global__ __launch_bounds__(32) void flash_attn_kernel(
    const unsigned short* __restrict__ Qb,
    const unsigned short* __restrict__ Kb,
    const unsigned short* __restrict__ Vb,
    unsigned short* __restrict__ AO)   // [B,T,E] bf16 (heads concatenated)
{
    const int lane = threadIdx.x;
    const int hi   = lane >> 4;
    const int lx   = lane & 15;

    int x = blockIdx.x;
    const int qblk = x & 127;  x >>= 7;      // T/16 = 128
    const int h    = x & 15;   x >>= 4;
    const int b    = x;
    const int q0   = qblk * 16;

    const size_t head = (((size_t)b * H_ + h) * T_) * D_;
    const unsigned short* Qp = Qb + head;
    const unsigned short* Kp = Kb + head;
    const unsigned short* Vp = Vb + head;

    __shared__ alignas(16) unsigned short Vt[64 * 32];   // [d][s_local]
    __shared__ alignas(16) unsigned short Kt[32 * 64];   // [s_local][d] raw copy
    __shared__ alignas(16) float          Srow[16 * 32]; // [row][s_local]
    __shared__ alignas(16) unsigned short Pbuf[16 * 32]; // [row][s_local] bf16
    __shared__ alignas(16) float          alphaB[16];
    __shared__ alignas(16) float          linvB[16];

    // Q fragments for d=[0,32) and d=[32,64)
    FragBf qf[2];
    #pragma unroll
    for (int f = 0; f < 2; ++f) {
        const unsigned short* p = Qp + (size_t)(q0 + lx) * D_ + f * 32 + (hi ? 8 : 0);
        qf[f].q[0] = *(const u32x4*)(p);
        qf[f].q[1] = *(const u32x4*)(p + 16);
    }

    v8f o[4];
    for (int nb = 0; nb < 4; ++nb) o[nb] = vzero8();
    float m = -1e30f, l = 0.0f;
    const int row = lx;
    const int qg  = q0 + row;

    for (int s0 = 0; s0 <= q0 + 15; s0 += 32) {
#if HAVE_ASYNC_LDS
        // ---- async bulk copy of K block [32 x 64] bf16 (4KB) into LDS.
        {
            unsigned short* gsrc =
                const_cast<unsigned short*>(Kp) + (size_t)s0 * D_ + lane * 8;
            unsigned short* ldst = &Kt[lane * 8];
            #pragma unroll
            for (int c = 0; c < 8; ++c) {
                __builtin_amdgcn_global_load_async_to_lds_b128(
                    (as1_v4i_ptr)(gsrc + c * 256),
                    (as3_v4i_ptr)(ldst + c * 256),
                    0, 0);
            }
        }
#endif
        // ---- V tile -> LDS transposed [d][s], packed b32 writes
        {
            const unsigned short* r0 = Vp + (size_t)(s0 + 2 * lx) * D_ + hi * 32;
            const unsigned short* r1 = r0 + D_;
            union { u32x4 q[4]; unsigned short h16[32]; } v0, v1;
            #pragma unroll
            for (int i = 0; i < 4; ++i) { v0.q[i] = ((const u32x4*)r0)[i]; v1.q[i] = ((const u32x4*)r1)[i]; }
            #pragma unroll
            for (int dd = 0; dd < 32; ++dd) {
                int d = hi * 32 + dd;
                *(unsigned int*)&Vt[d * 32 + 2 * lx] =
                    (unsigned int)v0.h16[dd] | ((unsigned int)v1.h16[dd] << 16);
            }
        }

#if HAVE_ASYNC_LDS
        asm volatile("s_wait_asynccnt 0x0" ::: "memory");
#endif
        // ---- S = Q * K^T : load all 4 K fragments first, then 4 WMMAs
        FragBf kf[4];
        #pragma unroll
        for (int sub = 0; sub < 2; ++sub)
            #pragma unroll
            for (int f = 0; f < 2; ++f) {
#if HAVE_ASYNC_LDS
                const unsigned short* p =
                    &Kt[(sub * 16 + lx) * 64 + f * 32 + (hi ? 8 : 0)];
#else
                const unsigned short* p =
                    Kp + (size_t)(s0 + sub * 16 + lx) * D_ + f * 32 + (hi ? 8 : 0);
#endif
                kf[sub * 2 + f].q[0] = *(const u32x4*)(p);
                kf[sub * 2 + f].q[1] = *(const u32x4*)(p + 16);
            }
        #pragma unroll
        for (int sub = 0; sub < 2; ++sub) {
            v8f sacc = vzero8();
            #pragma unroll
            for (int f = 0; f < 2; ++f)
                sacc = __builtin_amdgcn_wmma_f32_16x16x32_bf16(
                    false, qf[f].v, false, kf[sub * 2 + f].v,
                    (short)0, sacc, false, false);
            #pragma unroll
            for (int v = 0; v < 8; ++v)
                Srow[(v + 8 * hi) * 32 + sub * 16 + lx] = sacc[v];
        }
        __syncthreads();

        // ---- online softmax, per row (both half-waves compute identical copies)
        {
            f32x4 sv4[8];
            const f32x4* sr4 = (const f32x4*)&Srow[row * 32];
            #pragma unroll
            for (int i = 0; i < 8; ++i) sv4[i] = sr4[i];
            float sv[32];
            #pragma unroll
            for (int c = 0; c < 32; ++c) sv[c] = sv4[c >> 2][c & 3];

            float bm = -1e30f;
            #pragma unroll
            for (int c = 0; c < 32; ++c)
                if (s0 + c <= qg && sv[c] > bm) bm = sv[c];
            float mn    = (m > bm) ? m : bm;
            float alpha = __expf(m - mn);
            float ssum  = 0.0f;
            #pragma unroll
            for (int c = 0; c < 32; c += 2) {
                float p0 = (s0 + c     <= qg) ? __expf(sv[c]     - mn) : 0.0f;
                float p1 = (s0 + c + 1 <= qg) ? __expf(sv[c + 1] - mn) : 0.0f;
                ssum += p0 + p1;
                *(unsigned int*)&Pbuf[row * 32 + c] = pack_bf2(p0, p1);
            }
            l = l * alpha + ssum;
            m = mn;
            alphaB[row] = alpha;
        }
        __syncthreads();

        // ---- rescale O by alpha (broadcast via LDS)
        {
            f32x4 a0 = *(const f32x4*)&alphaB[8 * hi];
            f32x4 a1 = *(const f32x4*)&alphaB[8 * hi + 4];
            float av[8] = {a0.x, a0.y, a0.z, a0.w, a1.x, a1.y, a1.z, a1.w};
            #pragma unroll
            for (int nb = 0; nb < 4; ++nb)
                #pragma unroll
                for (int v = 0; v < 8; ++v) o[nb][v] *= av[v];
        }

        // ---- O += P * V : load P and all 4 V fragments first, then 4 WMMAs
        FragBf pf;
        {
            const unsigned short* p = &Pbuf[lx * 32 + (hi ? 8 : 0)];
            pf.q[0] = *(const u32x4*)(p);
            pf.q[1] = *(const u32x4*)(p + 16);
        }
        FragBf vf[4];
        #pragma unroll
        for (int nb = 0; nb < 4; ++nb) {
            const unsigned short* p = &Vt[(nb * 16 + lx) * 32 + (hi ? 8 : 0)];
            vf[nb].q[0] = *(const u32x4*)(p);
            vf[nb].q[1] = *(const u32x4*)(p + 16);
        }
        #pragma unroll
        for (int nb = 0; nb < 4; ++nb)
            o[nb] = __builtin_amdgcn_wmma_f32_16x16x32_bf16(
                false, pf.v, false, vf[nb].v, (short)0, o[nb], false, false);
        __syncthreads();
    }

    // ---- finalize: O /= l, store bf16 into AO[b, t, h*64 + d]
    linvB[row] = (l > 0.0f) ? (1.0f / l) : 0.0f;
    __syncthreads();
    f32x4 L0 = *(const f32x4*)&linvB[8 * hi];
    f32x4 L1 = *(const f32x4*)&linvB[8 * hi + 4];
    float lv[8] = {L0.x, L0.y, L0.z, L0.w, L1.x, L1.y, L1.z, L1.w};

    unsigned short* ao = AO + (size_t)b * T_ * E_;
    #pragma unroll
    for (int nb = 0; nb < 4; ++nb)
        #pragma unroll
        for (int v = 0; v < 8; ++v) {
            int t = q0 + v + 8 * hi;
            int e = h * 64 + nb * 16 + lx;
            ao[(size_t)t * E_ + e] = f2bf(o[nb][v] * lv[v]);
        }
}

// ---------------------------------------------------------------------------
// Kernel 3: output projection  out = AO(bf16) @ Wp + bp   (f32 out)
// Block = 128 threads (4 waves); 64 rows x 64 cols; 64-deep K chunks.
// grid.x = (B*T/64) * (E/64)
// ---------------------------------------------------------------------------
__global__ __launch_bounds__(128) void out_proj_kernel(
    const unsigned short* __restrict__ AO,
    const float* __restrict__ Wp, const float* __restrict__ bp,
    float* __restrict__ out)
{
    const int tid  = threadIdx.x;
    const int lane = tid & 31;
    const int wave = tid >> 5;
    const int hi   = lane >> 4;
    const int lx   = lane & 15;

    const int gx = blockIdx.x & 15;       // E/64 = 16 column chunks
    const int gy = blockIdx.x >> 4;       // B*T/64 = 128 row chunks
    const int n0 = gx * 64;
    const int r0 = gy * 64 + wave * 16;   // global row in [0, B*T)

    __shared__ alignas(16) unsigned short Wt[64 * 64];   // [n_local][e_local], 8KB

    v8f acc[4];
    for (int nb = 0; nb < 4; ++nb) acc[nb] = vzero8();

    const unsigned short* Arow = AO + (size_t)r0 * E_;

    for (int e0 = 0; e0 < E_; e0 += 64) {
        if (e0 + 64 < E_) __builtin_prefetch(Wp + (size_t)(e0 + 64) * E_ + n0, 0, 0);

        // Cooperative transpose-load of Wp[e0..e0+63][n0..n0+63]
        #pragma unroll
        for (int r = 0; r < 16; ++r) {
            int idx = tid + 128 * r;      // 0..2047 = n*32 + epair
            int n   = idx >> 5;
            int ep  = idx & 31;
            float w0 = Wp[(size_t)(e0 + 2 * ep)     * E_ + n0 + n];
            float w1 = Wp[(size_t)(e0 + 2 * ep + 1) * E_ + n0 + n];
            *(unsigned int*)&Wt[n * 64 + 2 * ep] = pack_bf2(w0, w1);
        }
        __syncthreads();

        // A fragments directly from bf16 AO
        FragBf a[2];
        #pragma unroll
        for (int half = 0; half < 2; ++half) {
            const unsigned short* p =
                Arow + (size_t)lx * E_ + e0 + half * 32 + (hi ? 8 : 0);
            a[half].q[0] = *(const u32x4*)(p);
            a[half].q[1] = *(const u32x4*)(p + 16);
        }

        // Load ALL B fragments, then 8 WMMAs back-to-back
        FragBf bf[8];
        #pragma unroll
        for (int half = 0; half < 2; ++half)
            #pragma unroll
            for (int nb = 0; nb < 4; ++nb) {
                const unsigned short* p =
                    &Wt[(nb * 16 + lx) * 64 + half * 32 + (hi ? 8 : 0)];
                bf[half * 4 + nb].q[0] = *(const u32x4*)(p);
                bf[half * 4 + nb].q[1] = *(const u32x4*)(p + 16);
            }
        #pragma unroll
        for (int half = 0; half < 2; ++half)
            #pragma unroll
            for (int nb = 0; nb < 4; ++nb)
                acc[nb] = __builtin_amdgcn_wmma_f32_16x16x32_bf16(
                    false, a[half].v, false, bf[half * 4 + nb].v,
                    (short)0, acc[nb], false, false);
        __syncthreads();
    }

    float bias[4];
    #pragma unroll
    for (int nb = 0; nb < 4; ++nb) bias[nb] = bp[n0 + nb * 16 + lx];

    #pragma unroll
    for (int nb = 0; nb < 4; ++nb)
        #pragma unroll
        for (int v = 0; v < 8; ++v) {
            size_t rrow = (size_t)(r0 + v + 8 * hi);
            out[rrow * E_ + n0 + nb * 16 + lx] = acc[nb][v] + bias[nb];
        }
}

// ---------------------------------------------------------------------------
// Host launcher.  Workspace: 4 bf16 buffers of B*H*T*D (=B*T*E) elems = 64 MB.
// ---------------------------------------------------------------------------
extern "C" void kernel_launch(void* const* d_in, const int* in_sizes, int n_in,
                              void* d_out, int out_size, void* d_ws, size_t ws_size,
                              hipStream_t stream)
{
    const float* data = (const float*)d_in[0];
    const float* Wq   = (const float*)d_in[1];
    const float* Wk   = (const float*)d_in[2];
    const float* Wv   = (const float*)d_in[3];
    const float* Wp   = (const float*)d_in[4];
    const float* bp   = (const float*)d_in[5];
    float* out        = (float*)d_out;

    const size_t N = (size_t)B_ * H_ * T_ * D_;   // 8,388,608 elements
    unsigned short* Qb = (unsigned short*)d_ws;
    unsigned short* Kb = Qb + N;
    unsigned short* Vb = Kb + N;
    unsigned short* AO = Vb + N;

    // 1) QKV projections -> bf16 [B,H,T,D] (Q pre-scaled by 1/sqrt(D))
    qkv_proj_kernel<<<3 * B_ * H_ * (T_ / 64), 128, 0, stream>>>(
        data, Wq, Wk, Wv, Qb, Kb, Vb);

    // 2) causal flash attention -> bf16 AO [B,T,E]
    flash_attn_kernel<<<B_ * H_ * (T_ / 16), 32, 0, stream>>>(Qb, Kb, Vb, AO);

    // 3) output projection + bias -> f32 out [B,T,E]
    out_proj_kernel<<<(B_ * T_ / 64) * (E_ / 64), 128, 0, stream>>>(
        AO, Wp, bp, out);
}
